// SwinUNETR_46935402610973
// MI455X (gfx1250) — matrix operand order repaired and checked by reading, not compiled, not run on hardware
//
#include <hip/hip_runtime.h>
#include <hip/hip_bf16.h>

typedef __attribute__((ext_vector_type(8)))  __bf16 v8bf;
typedef __attribute__((ext_vector_type(16))) __bf16 v16bf;
typedef __attribute__((ext_vector_type(8)))  float  v8f;

// ---------------------------------------------------------------------------
// WMMA fragment helpers (CDNA5 16x16x32 bf16, wave32)
// A (16x32 bf16): lanes0-15 row=lane, halves = K{0..7},{16..23}; lanes16-31 row=lane-16, K{8..15},{24..31}
// B (32x16 bf16): lanes0-15 col=lane, K{0..7},{16..23}; lanes16-31 col=lane-16, K{8..15},{24..31}
// C/D (16x16 f32): vgpr i: lanes0-15 -> M=i, lanes16-31 -> M=i+8; N=lane&15
// ---------------------------------------------------------------------------
__device__ __forceinline__ v16bf frag_a(const __bf16* base, int ld) {
  int lane = threadIdx.x & 31;
  int r = lane & 15, hi = lane >> 4;
  const __bf16* p = base + (size_t)r * ld + hi * 8;
  v8bf lo = *(const v8bf*)p;
  v8bf hh = *(const v8bf*)(p + 16);
  v16bf f;
#pragma unroll
  for (int i = 0; i < 8; ++i) { f[i] = lo[i]; f[i + 8] = hh[i]; }
  return f;
}

// A-frag with row clamp (for attention Q tiles that straddle the 216 boundary)
__device__ __forceinline__ v16bf frag_a_rows(const __bf16* mat, int row0, int maxrow, int ld) {
  int lane = threadIdx.x & 31;
  int r = lane & 15, hi = lane >> 4;
  int row = row0 + r; if (row > maxrow) row = maxrow;
  const __bf16* p = mat + (size_t)row * ld + hi * 8;
  v8bf lo = *(const v8bf*)p;
  v8bf hh = *(const v8bf*)(p + 16);
  v16bf f;
#pragma unroll
  for (int i = 0; i < 8; ++i) { f[i] = lo[i]; f[i + 8] = hh[i]; }
  return f;
}

__device__ __forceinline__ v16bf frag_b(const __bf16* base, int ld) {
  int lane = threadIdx.x & 31;
  int n = lane & 15, hi = lane >> 4;
  const __bf16* p = base + (size_t)n * ld + hi * 8;
  v8bf a0 = *(const v8bf*)p;
  v8bf a1 = *(const v8bf*)(p + 16);
  v16bf f;
#pragma unroll
  for (int i = 0; i < 8; ++i) { f[i] = a0[i]; f[i + 8] = a1[i]; }
  return f;
}

__device__ __forceinline__ v8f wmma_bf16(v16bf a, v16bf b, v8f c) {
  return __builtin_amdgcn_wmma_f32_16x16x32_bf16(false, a, false, b, (short)0, c, false, false);
}

// ---------------------------------------------------------------------------
// CDNA5 async global->LDS copy (ASYNCcnt-tracked, no VGPR round-trip).
// CDNA5 ISA / 15.18.3: GLOBAL_LOAD_ASYNC_TO_LDS_B128, GV mode:
//   each lane copies 16 bytes from its global address to its LDS address.
// LDS address = low 32 bits of the generic pointer (wave-relative).
// ---------------------------------------------------------------------------
__device__ __forceinline__ void async_g2l_b128(void* lds_ptr, const void* gptr) {
  unsigned int loff = (unsigned int)(unsigned long long)lds_ptr;
  unsigned long long ga = (unsigned long long)gptr;
  asm volatile("global_load_async_to_lds_b128 %0, %1, off"
               :: "v"(loff), "v"(ga)
               : "memory");
}

__device__ __forceinline__ void wait_async0() {
#if defined(__has_builtin)
#if __has_builtin(__builtin_amdgcn_s_wait_asynccnt)
  __builtin_amdgcn_s_wait_asynccnt(0);
#else
  asm volatile("s_wait_asynccnt 0x0" ::: "memory");
#endif
#else
  asm volatile("s_wait_asynccnt 0x0" ::: "memory");
#endif
}

// ---------------------------------------------------------------------------
// elementwise / small kernels
// ---------------------------------------------------------------------------
__global__ void k_cvt(const float* __restrict__ s, __bf16* __restrict__ d, int n) {
  int i = blockIdx.x * 256 + threadIdx.x;
  if (i < n) d[i] = (__bf16)s[i];
}

// patch embed: Conv3d(1->96, k=2, s=2) over 96^3 input -> tokens (48^3, 96)
__global__ void k_conv(const float* __restrict__ x, const float* __restrict__ w,
                       const float* __restrict__ b, float* __restrict__ T) {
  int idx = blockIdx.x * 256 + threadIdx.x;
  if (idx >= 48 * 48 * 48 * 96) return;
  int c = idx % 96, l = idx / 96;
  int wz = l % 48, wy = (l / 48) % 48, wx = l / (48 * 48);
  const float* xs = x + ((size_t)(wx * 2) * 96 + wy * 2) * 96 + wz * 2;
  const float* wc = w + c * 8;
  float acc = b[c];
#pragma unroll
  for (int kd = 0; kd < 2; ++kd)
#pragma unroll
    for (int kh = 0; kh < 2; ++kh)
#pragma unroll
      for (int kw = 0; kw < 2; ++kw)
        acc += xs[kd * 9216 + kh * 96 + kw] * wc[kd * 4 + kh * 2 + kw];
  T[idx] = acc;
}

// relative position bias table -> (heads, 216, 216)
__global__ void k_bias(const float* __restrict__ rpb, float* __restrict__ Bias, int heads) {
  int idx = blockIdx.x * 256 + threadIdx.x;
  if (idx >= heads * 216 * 216) return;
  int m = idx % 216, nn = (idx / 216) % 216, h = idx / (216 * 216);
  int i0 = nn / 36, i1 = (nn / 6) % 6, i2 = nn % 6;
  int j0 = m / 36, j1 = (m / 6) % 6, j2 = m % 6;
  int t = (i0 - j0 + 5) * 121 + (i1 - j1 + 5) * 11 + (i2 - j2 + 5);
  Bias[idx] = rpb[t * heads + h];
}

// LayerNorm (wave per row), optional shifted window-partition permutation on output
__global__ void k_ln(const float* __restrict__ T, const float* __restrict__ g,
                     const float* __restrict__ bt, __bf16* __restrict__ X,
                     int L, int C, int S, int shift, int part) {
  int row = blockIdx.x * 8 + (threadIdx.x >> 5);
  int lane = threadIdx.x & 31;
  if (row >= L) return;
  const float* tr = T + (size_t)row * C;
  float loc[6];
  float s = 0.f;
#pragma unroll
  for (int t = 0; t < 6; ++t) {
    int i = lane + t * 32;
    float v = (i < C) ? tr[i] : 0.f;
    loc[t] = v; s += v;
  }
#pragma unroll
  for (int m = 1; m <= 16; m <<= 1) s += __shfl_xor(s, m);
  float mu = s / C;
  float vv = 0.f;
#pragma unroll
  for (int t = 0; t < 6; ++t) {
    int i = lane + t * 32;
    if (i < C) { float d = loc[t] - mu; vv += d * d; }
  }
#pragma unroll
  for (int m = 1; m <= 16; m <<= 1) vv += __shfl_xor(vv, m);
  float inv = rsqrtf(vv / C + 1e-5f);
  int r = row;
  if (part) {
    int W6 = S / 6;
    int a = row / (S * S), b2 = (row / S) % S, c2 = row % S;
    int pa = (a - shift + S) % S, pb = (b2 - shift + S) % S, pc = (c2 - shift + S) % S;
    int widx = ((pa / 6) * W6 + (pb / 6)) * W6 + (pc / 6);
    int n = (pa % 6) * 36 + (pb % 6) * 6 + (pc % 6);
    r = widx * 216 + n;
  }
  __bf16* xr = X + (size_t)r * C;
#pragma unroll
  for (int t = 0; t < 6; ++t) {
    int i = lane + t * 32;
    if (i < C) xr[i] = (__bf16)((loc[t] - mu) * inv * g[i] + bt[i]);
  }
}

// Patch-merge gather (8-neighbor concat, C 96->768) + LayerNorm -> bf16 rows (13824 x 768)
__global__ void k_ln_merge(const float* __restrict__ T, const float* __restrict__ g,
                           const float* __restrict__ bt, __bf16* __restrict__ X) {
  int row = blockIdx.x * 8 + (threadIdx.x >> 5);
  int lane = threadIdx.x & 31;
  if (row >= 13824) return;
  const int oi[8] = {0, 1, 0, 0, 1, 1, 0, 1};
  const int oj[8] = {0, 0, 1, 0, 1, 0, 1, 1};
  const int ok[8] = {0, 0, 0, 1, 0, 1, 1, 1};
  int a2 = row / 576, b2 = (row / 24) % 24, c2 = row % 24;
  float vals[24];
  float s = 0.f;
#pragma unroll
  for (int t = 0; t < 24; ++t) {
    int col = lane + t * 32;
    int o = col / 96, c = col % 96;
    int l = ((2 * a2 + oi[o]) * 48 + (2 * b2 + oj[o])) * 48 + (2 * c2 + ok[o]);
    float v = T[(size_t)l * 96 + c];
    vals[t] = v; s += v;
  }
#pragma unroll
  for (int m = 1; m <= 16; m <<= 1) s += __shfl_xor(s, m);
  float mu = s / 768.f;
  float vv = 0.f;
#pragma unroll
  for (int t = 0; t < 24; ++t) { float d = vals[t] - mu; vv += d * d; }
#pragma unroll
  for (int m = 1; m <= 16; m <<= 1) vv += __shfl_xor(vv, m);
  float inv = rsqrtf(vv / 768.f + 1e-5f);
  __bf16* xr = X + (size_t)row * 768;
#pragma unroll
  for (int t = 0; t < 24; ++t) {
    int col = lane + t * 32;
    xr[col] = (__bf16)((vals[t] - mu) * inv * g[col] + bt[col]);
  }
}

// window-reverse residual add: T(token order) += P(window order)
__global__ void k_res_win(float* __restrict__ T, const float* __restrict__ P,
                          int L, int C, int S, int shift) {
  int idx = blockIdx.x * 256 + threadIdx.x;
  if (idx >= L * C) return;
  int c = idx % C, l = idx / C;
  int W6 = S / 6;
  int a = l / (S * S), b2 = (l / S) % S, c2 = l % S;
  int pa = (a - shift + S) % S, pb = (b2 - shift + S) % S, pc = (c2 - shift + S) % S;
  int widx = ((pa / 6) * W6 + (pb / 6)) * W6 + (pc / 6);
  int n = (pa % 6) * 36 + (pb % 6) * 6 + (pc % 6);
  T[idx] += P[(size_t)(widx * 216 + n) * C + c];
}

__global__ void k_add(float* __restrict__ T, const float* __restrict__ Y, int n) {
  int i = blockIdx.x * 256 + threadIdx.x;
  if (i < n) T[i] += Y[i];
}
__global__ void k_add_out(const float* __restrict__ T, const float* __restrict__ Y,
                          float* __restrict__ O, int n) {
  int i = blockIdx.x * 256 + threadIdx.x;
  if (i < n) O[i] = T[i] + Y[i];
}

// ---------------------------------------------------------------------------
// Generic WMMA GEMM: out[M,N] = A[M,K](bf16) * W[N,K]^T (+bias) ; flags:
//   bit0: bf16 output, bit1: exact GELU, bit2: scale cols<qC by 1/sqrt(32)
// grid = (M/128, N/32), 256 threads (8 waves: wave -> 16 rows x 32 cols)
// ---------------------------------------------------------------------------
__global__ void __launch_bounds__(256) k_gemm(const __bf16* __restrict__ A,
                                              const __bf16* __restrict__ W,
                                              const float* __restrict__ bias,
                                              void* __restrict__ out,
                                              int M, int N, int K, int flags, int qC) {
  int wv = threadIdx.x >> 5;
  int row0 = blockIdx.x * 128 + wv * 16;
  int col0 = blockIdx.y * 32;
  v8f acc0 = (v8f)0.0f, acc1 = (v8f)0.0f;
  const __bf16* Ar = A + (size_t)row0 * K;
  const __bf16* W0 = W + (size_t)col0 * K;
  const __bf16* W1 = W + (size_t)(col0 + 16) * K;
  for (int k0 = 0; k0 < K; k0 += 32) {
    v16bf a  = frag_a(Ar + k0, K);
    v16bf b0 = frag_b(W0 + k0, K);
    v16bf b1 = frag_b(W1 + k0, K);
    acc0 = wmma_bf16(a, b0, acc0);
    acc1 = wmma_bf16(a, b1, acc1);
  }
  int lane = threadIdx.x & 31, coll = lane & 15, hi = lane >> 4;
#pragma unroll
  for (int t = 0; t < 2; ++t) {
    v8f acc = t ? acc1 : acc0;
    int colg = col0 + t * 16 + coll;
    float bb = bias ? bias[colg] : 0.f;
#pragma unroll
    for (int i = 0; i < 8; ++i) {
      int rowg = row0 + i + hi * 8;
      float v = acc[i] + bb;
      if (flags & 2) v = 0.5f * v * (1.f + erff(v * 0.70710678118f));
      if ((flags & 4) && colg < qC) v *= 0.17677669529663687f;  // 1/sqrt(hd=32)
      size_t o = (size_t)rowg * N + colg;
      if (flags & 1) ((__bf16*)out)[o] = (__bf16)v;
      else           ((float*)out)[o]  = v;
    }
  }
}

// ---------------------------------------------------------------------------
// Windowed attention: one block per (window, head). N=216 (pad 224), hd=32.
// 7 waves; wave handles row tiles rt and rt+7. LDS: K(224x32) Vt(32x224) +
// per-wave 16x32 P staging = 35,840 B dynamic.
// K rows staged via GLOBAL_LOAD_ASYNC_TO_LDS_B128 (ASYNCcnt path).
// ---------------------------------------------------------------------------
__global__ void __launch_bounds__(224) k_attn(const __bf16* __restrict__ QKV,
                                              const float* __restrict__ Bias,
                                              __bf16* __restrict__ Aout,
                                              int heads, int C) {
  extern __shared__ __bf16 smem[];
  __bf16* Ks = smem;             // 224*32
  __bf16* Vt = Ks + 224 * 32;    // 32*224 (transposed V)
  __bf16* Ps = Vt + 32 * 224;    // 7 waves * 16*32
  int w = blockIdx.x / heads, h = blockIdx.x % heads;
  int tid = threadIdx.x, lane = tid & 31, wv = tid >> 5;
  int C3 = 3 * C;
  const __bf16* base = QKV + (size_t)w * 216 * C3 + h * 32;  // q at +0, k at +C, v at +2C
  int n = tid;
  if (n < 216) {
    const __bf16* src = base + (size_t)n * C3;
    // K row: 64B = 4 async 16B copies straight into LDS
#pragma unroll
    for (int j = 0; j < 4; ++j)
      async_g2l_b128(Ks + n * 32 + j * 8, src + C + j * 8);
    // V row: load + transpose into Vt
#pragma unroll
    for (int d = 0; d < 32; ++d)
      Vt[d * 224 + n] = src[2 * C + d];
  } else {
#pragma unroll
    for (int d = 0; d < 32; ++d) {
      Ks[n * 32 + d]  = (__bf16)0.f;
      Vt[d * 224 + n] = (__bf16)0.f;
    }
  }
  wait_async0();
  __syncthreads();
  const float* Bh = Bias + (size_t)h * 216 * 216;
  int coll = lane & 15, hi = lane >> 4;
  for (int rt = wv; rt < 14; rt += 7) {
    int row0 = rt * 16;
    v16bf qa = frag_a_rows(base, row0, 215, C3);  // q is already scaled in qkv GEMM
    v8f sc[14];
#pragma unroll
    for (int ct = 0; ct < 14; ++ct) {
      v16bf kb = frag_b(Ks + ct * 16 * 32, 32);
      sc[ct] = wmma_bf16(qa, kb, (v8f)0.0f);
    }
    float mx[8], sm[8];
#pragma unroll
    for (int i = 0; i < 8; ++i) mx[i] = -3.0e38f;
#pragma unroll
    for (int ct = 0; ct < 14; ++ct) {
      int ccol = ct * 16 + coll;
#pragma unroll
      for (int i = 0; i < 8; ++i) {
        int rrow = row0 + i + hi * 8;
        float v = sc[ct][i];
        if (ccol < 216) { if (rrow < 216) v += Bh[rrow * 216 + ccol]; }
        else v = -3.0e38f;
        sc[ct][i] = v;
        mx[i] = fmaxf(mx[i], v);
      }
    }
#pragma unroll
    for (int i = 0; i < 8; ++i) {
      float m = mx[i];
      m = fmaxf(m, __shfl_xor(m, 1));
      m = fmaxf(m, __shfl_xor(m, 2));
      m = fmaxf(m, __shfl_xor(m, 4));
      m = fmaxf(m, __shfl_xor(m, 8));
      mx[i] = m; sm[i] = 0.f;
    }
#pragma unroll
    for (int ct = 0; ct < 14; ++ct)
#pragma unroll
      for (int i = 0; i < 8; ++i) {
        float e = __expf(sc[ct][i] - mx[i]);
        sc[ct][i] = e; sm[i] += e;
      }
#pragma unroll
    for (int i = 0; i < 8; ++i) {
      float s2 = sm[i];
      s2 += __shfl_xor(s2, 1);
      s2 += __shfl_xor(s2, 2);
      s2 += __shfl_xor(s2, 4);
      s2 += __shfl_xor(s2, 8);
      sm[i] = 1.f / s2;
    }
    // O = P @ V, streaming P through a per-wave 16x32 LDS tile
    __bf16* Pm = Ps + wv * 16 * 32;
    v8f o0 = (v8f)0.0f, o1 = (v8f)0.0f;
#pragma unroll
    for (int kc = 0; kc < 7; ++kc) {
#pragma unroll
      for (int u = 0; u < 2; ++u) {
        int ct = kc * 2 + u;
#pragma unroll
        for (int i = 0; i < 8; ++i)
          Pm[(i + hi * 8) * 32 + u * 16 + coll] = (__bf16)(sc[ct][i] * sm[i]);
      }
      v16bf pa  = frag_a(Pm, 32);
      v16bf vb0 = frag_b(Vt + kc * 32, 224);
      v16bf vb1 = frag_b(Vt + 16 * 224 + kc * 32, 224);
      o0 = wmma_bf16(pa, vb0, o0);
      o1 = wmma_bf16(pa, vb1, o1);
    }
#pragma unroll
    for (int i = 0; i < 8; ++i) {
      int rrow = row0 + i + hi * 8;
      if (rrow < 216) {
        size_t orow = ((size_t)w * 216 + rrow) * C + h * 32;
        Aout[orow + coll]      = (__bf16)o0[i];
        Aout[orow + 16 + coll] = (__bf16)o1[i];
      }
    }
  }
}

// ---------------------------------------------------------------------------
// host orchestration
// ---------------------------------------------------------------------------
extern "C" void kernel_launch(void* const* d_in, const int* in_sizes, int n_in,
                              void* d_out, int out_size, void* d_ws, size_t ws_size,
                              hipStream_t stream) {
  const float* x      = (const float*)d_in[0];
  const float* conv_w = (const float*)d_in[1];
  const float* conv_b = (const float*)d_in[2];
  auto P = [&](int i) { return (const float*)d_in[i]; };
  // block param base indices: blk1=3, blk2=16, merge=29, blk3=32
  // within block: 0 n1_g 1 n1_b 2 qkv_w 3 qkv_b 4 rpb 5 proj_w 6 proj_b
  //               7 n2_g 8 n2_b 9 fc1_w 10 fc1_b 11 fc2_w 12 fc2_b

  char* ws = (char*)d_ws;
  size_t off = 0;
  auto take = [&](size_t bytes) -> char* {
    char* p = ws + off;
    off += (bytes + 255) & ~(size_t)255;
    return p;
  };
  const int L1 = 110592, L3 = 13824;
  float*  T0   = (float*)take((size_t)L1 * 96 * 4);
  __bf16* Xst  = (__bf16*)take((size_t)L1 * 96 * 2);          // also holds 13824x768
  __bf16* BIG  = (__bf16*)take((size_t)L1 * 384 * 2);         // qkv / fc1 staging
  __bf16* AO   = (__bf16*)take((size_t)L1 * 96 * 2);
  float*  tmpF = (float*)take((size_t)L1 * 96 * 4);
  float*  T2   = (float*)take((size_t)L3 * 192 * 4);
  float*  BiasB= (float*)take((size_t)6 * 216 * 216 * 4);
  __bf16* wq1  = (__bf16*)take(27648 * 2);
  __bf16* wp1  = (__bf16*)take(9216 * 2);
  __bf16* wf11 = (__bf16*)take(36864 * 2);
  __bf16* wf21 = (__bf16*)take(36864 * 2);
  __bf16* wq2  = (__bf16*)take(27648 * 2);
  __bf16* wp2  = (__bf16*)take(9216 * 2);
  __bf16* wf12 = (__bf16*)take(36864 * 2);
  __bf16* wf22 = (__bf16*)take(36864 * 2);
  __bf16* wq3  = (__bf16*)take(110592 * 2);
  __bf16* wp3  = (__bf16*)take(36864 * 2);
  __bf16* wf13 = (__bf16*)take(147456 * 2);
  __bf16* wf23 = (__bf16*)take(147456 * 2);
  __bf16* wred = (__bf16*)take(147456 * 2);

  auto cvt = [&](const float* s, __bf16* d, int n) {
    k_cvt<<<(n + 255) / 256, 256, 0, stream>>>(s, d, n);
  };
  cvt(P(3 + 2), wq1, 27648);   cvt(P(3 + 5), wp1, 9216);
  cvt(P(3 + 9), wf11, 36864);  cvt(P(3 + 11), wf21, 36864);
  cvt(P(16 + 2), wq2, 27648);  cvt(P(16 + 5), wp2, 9216);
  cvt(P(16 + 9), wf12, 36864); cvt(P(16 + 11), wf22, 36864);
  cvt(P(32 + 2), wq3, 110592); cvt(P(32 + 5), wp3, 36864);
  cvt(P(32 + 9), wf13, 147456);cvt(P(32 + 11), wf23, 147456);
  cvt(P(31), wred, 147456);

  // patch embed
  k_conv<<<(L1 * 96 + 255) / 256, 256, 0, stream>>>(x, conv_w, conv_b, T0);

  auto run_block = [&](int pb, float* T, int L, int S, int C, int heads, int shift,
                       __bf16* wq, __bf16* wp, __bf16* wf1, __bf16* wf2,
                       float* finalOut) {
    int nW = (S / 6) * (S / 6) * (S / 6);
    int nb = heads * 216 * 216;
    k_bias<<<(nb + 255) / 256, 256, 0, stream>>>(P(pb + 4), BiasB, heads);
    k_ln<<<L / 8, 256, 0, stream>>>(T, P(pb + 0), P(pb + 1), Xst, L, C, S, shift, 1);
    dim3 g1(L / 128, (3 * C) / 32);
    k_gemm<<<g1, 256, 0, stream>>>(Xst, wq, P(pb + 3), BIG, L, 3 * C, C, 1 | 4, C);
    k_attn<<<nW * heads, 224, 35840, stream>>>(BIG, BiasB, AO, heads, C);
    dim3 g2(L / 128, C / 32);
    k_gemm<<<g2, 256, 0, stream>>>(AO, wp, P(pb + 6), tmpF, L, C, C, 0, 0);
    k_res_win<<<(L * C + 255) / 256, 256, 0, stream>>>(T, tmpF, L, C, S, shift);
    k_ln<<<L / 8, 256, 0, stream>>>(T, P(pb + 7), P(pb + 8), Xst, L, C, S, 0, 0);
    dim3 g3(L / 128, (4 * C) / 32);
    k_gemm<<<g3, 256, 0, stream>>>(Xst, wf1, P(pb + 10), BIG, L, 4 * C, C, 1 | 2, 0);
    dim3 g4(L / 128, C / 32);
    k_gemm<<<g4, 256, 0, stream>>>(BIG, wf2, P(pb + 12), tmpF, L, C, 4 * C, 0, 0);
    if (finalOut)
      k_add_out<<<(L * C + 255) / 256, 256, 0, stream>>>(T, tmpF, finalOut, L * C);
    else
      k_add<<<(L * C + 255) / 256, 256, 0, stream>>>(T, tmpF, L * C);
  };

  run_block(3,  T0, L1, 48, 96, 3, 0, wq1, wp1, wf11, wf21, nullptr);
  run_block(16, T0, L1, 48, 96, 3, 3, wq2, wp2, wf12, wf22, nullptr);

  // patch merge: gather+LN (768) then reduction GEMM to (13824, 192)
  k_ln_merge<<<L3 / 8, 256, 0, stream>>>(T0, P(29), P(30), Xst);
  dim3 gm(L3 / 128, 192 / 32);
  k_gemm<<<gm, 256, 0, stream>>>(Xst, wred, nullptr, T2, L3, 192, 768, 0, 0);

  run_block(32, T2, L3, 24, 192, 6, 0, wq3, wp3, wf13, wf23, (float*)d_out);
}
